// FlashDecoderLayer_39359080300891
// MI455X (gfx1250) — matrix operand order repaired and checked by reading, not compiled, not run on hardware
//
#include <hip/hip_runtime.h>
#include <hip/hip_bf16.h>

// ---------------- problem constants ----------------
constexpr int T_   = 2048;
constexpr int H_   = 4096;
constexpr int NH_  = 32;
constexpr int HS_  = 128;
constexpr int I_   = 13696;
constexpr int ROT_ = 32;

typedef __attribute__((ext_vector_type(16))) __bf16 v16bf;
typedef __attribute__((ext_vector_type(8)))  float  v8f;
typedef __attribute__((ext_vector_type(4)))  int    v4i;

// Async global->LDS path (gfx1250): guarded so a missing builtin falls back
// to the synchronous register-staged copy instead of breaking the compile.
#if defined(__has_builtin)
#  if __has_builtin(__builtin_amdgcn_global_load_async_to_lds_b128)
#    define USE_ASYNC_LDS 1
#  endif
#endif
#ifndef USE_ASYNC_LDS
#  define USE_ASYNC_LDS 0
#endif

#if USE_ASYNC_LDS
typedef __attribute__((address_space(1))) v4i ga_v4i;   // global int4*
typedef __attribute__((address_space(3))) v4i ls_v4i;   // LDS int4*
__device__ __forceinline__ void async_copy_32B(char* lptr, const __bf16* gptr) {
    __builtin_amdgcn_global_load_async_to_lds_b128(
        (ga_v4i*)gptr, (ls_v4i*)lptr, 0, 0);
    __builtin_amdgcn_global_load_async_to_lds_b128(
        (ga_v4i*)(gptr + 8), (ls_v4i*)(lptr + 16), 0, 0);
}
#endif

union FragU { uint4 u[2]; v16bf v; };

// A-operand frag (16x32 bf16): lane%16 = M row, half=lane/16.
// VGPRs 0-3 hold K[half*8 .. +8), VGPRs 4-7 hold K[16+half*8 .. +8).
__device__ __forceinline__ v16bf ldfragA(const char* rowbase, int half) {
    FragU f;
    f.u[0] = *(const uint4*)(rowbase + half * 16);
    f.u[1] = *(const uint4*)(rowbase + 32 + half * 16);
    return f.v;
}

// B-operand frag (32x16 bf16): lane%16 = N col, half=lane/16.
// VGPRs 0-3 hold K[half*8 .. +8), VGPRs 4-7 hold K[16+half*8 .. +8).
__device__ __forceinline__ v16bf ldfragB(const char* rowbase, int half) {
    FragU f;
    f.u[0] = *(const uint4*)(rowbase + half * 16);
    f.u[1] = *(const uint4*)(rowbase + 32 + half * 16);
    return f.v;
}

__device__ __forceinline__ v8f wmma_bf16(v16bf a, v16bf b, v8f c) {
    return __builtin_amdgcn_wmma_f32_16x16x32_bf16(
        false, a, false, b, (short)0, c, false, false);
}

// ---------------- RMSNorm (fused residual add) ----------------
__global__ __launch_bounds__(256)
void rmsnorm_kernel(const float* __restrict__ x, const float* __restrict__ addin,
                    const float* __restrict__ w, float* __restrict__ res_out,
                    __bf16* __restrict__ normed) {
    __shared__ float red[256];
    int t = blockIdx.x, tid = threadIdx.x;
    const float* xr = x + (size_t)t * H_;
    const float* ar = addin + (size_t)t * H_;
    float vals[16];
    float ss = 0.f;
#pragma unroll
    for (int j = 0; j < 16; ++j) {
        int c = tid + j * 256;
        float v = xr[c] + ar[c];
        vals[j] = v;
        res_out[(size_t)t * H_ + c] = v;
        ss += v * v;
    }
    red[tid] = ss;
    __syncthreads();
    for (int s = 128; s > 0; s >>= 1) {
        if (tid < s) red[tid] += red[tid + s];
        __syncthreads();
    }
    float r = rsqrtf(red[0] / (float)H_ + 1e-5f);
#pragma unroll
    for (int j = 0; j < 16; ++j) {
        int c = tid + j * 256;
        normed[(size_t)t * H_ + c] = (__bf16)(w[c] * vals[j] * r);
    }
}

// ---------------- generic WMMA GEMM: C[M,N] = A[M,K] * W[N,K]^T ----------------
// Software-pipelined; triple-buffered LDS => one barrier per K-step.
struct EpiF32 {
    float* C;
    const float* bias;   // nullable
    __device__ void operator()(int m, int n, int N, float v) const {
        C[(size_t)m * N + n] = bias ? (v + bias[n]) : v;
    }
};
struct EpiB16 {
    __bf16* C;
    __device__ void operator()(int m, int n, int N, float v) const {
        C[(size_t)m * N + n] = (__bf16)v;
    }
};

template <class Epi>
__global__ __launch_bounds__(256)
void gemm_kernel(const __bf16* __restrict__ A, const float* __restrict__ W,
                 int M, int N, int K, Epi epi) {
    __shared__ __align__(16) char sA[3][128 * 64];   // 128 rows x 32 bf16, x3 buffers
    __shared__ __align__(16) char sB[3][128 * 64];
    int m0 = blockIdx.y * 128, n0 = blockIdx.x * 128;
    int tid = threadIdx.x;
    int wave = tid >> 5, lane = tid & 31, half = lane >> 4, l16 = lane & 15;
    int wm = wave >> 2, wn = wave & 3;   // wave tile: 64(M) x 32(N)

    v8f acc[4][2] = {};

    int lrow = tid >> 1;
    int lcb  = (tid & 1) * 16;           // 16 elements along K per thread
    const __bf16* aptr = A + (size_t)(m0 + lrow) * K + lcb;
    const float*  wptr = W + (size_t)(n0 + lrow) * K + lcb;
    int soff = lrow * 64 + lcb * 2;
    int nkt = K >> 5;

    // ---- prologue: stage tile 0 ----
    float4 wr[4];
#pragma unroll
    for (int e = 0; e < 4; ++e) wr[e] = *(const float4*)(wptr + 4 * e);
#if USE_ASYNC_LDS
    async_copy_32B(sA[0] + soff, aptr);
#else
    uint4 ar0 = *(const uint4*)(aptr);
    uint4 ar1 = *(const uint4*)(aptr + 8);
#endif

    for (int kt = 0; kt < nkt; ++kt) {
        int b = kt % 3, bn = (kt + 1) % 3;
        bool more = (kt + 1 < nkt);

        // issue next tile's loads (overlaps with conversion + WMMA below)
        float4 wr2[4] = {wr[0], wr[1], wr[2], wr[3]};
#if USE_ASYNC_LDS
        if (more) async_copy_32B(sA[bn] + soff, aptr + 32);
#else
        uint4 an0 = ar0, an1 = ar1;
        if (more) { an0 = *(const uint4*)(aptr + 32); an1 = *(const uint4*)(aptr + 40); }
#endif
        if (more) {
#pragma unroll
            for (int e = 0; e < 4; ++e) wr2[e] = *(const float4*)(wptr + 32 + 4 * e);
            __builtin_prefetch(wptr + 64, 0, 1);   // global_prefetch_b8
        }

        // convert this tile's weights f32 -> bf16 and commit to LDS
        union { __bf16 h[16]; uint4 u[2]; } wb;
        const float wf[16] = {wr[0].x, wr[0].y, wr[0].z, wr[0].w,
                              wr[1].x, wr[1].y, wr[1].z, wr[1].w,
                              wr[2].x, wr[2].y, wr[2].z, wr[2].w,
                              wr[3].x, wr[3].y, wr[3].z, wr[3].w};
#pragma unroll
        for (int e = 0; e < 16; ++e) wb.h[e] = (__bf16)wf[e];
        *(uint4*)(sB[b] + soff)      = wb.u[0];
        *(uint4*)(sB[b] + soff + 16) = wb.u[1];
#if USE_ASYNC_LDS
        // tile kt's 2 async ops done when <=2 (tile kt+1's) remain (in-order)
        if (more) asm volatile("s_wait_asynccnt 2" ::: "memory");
        else      asm volatile("s_wait_asynccnt 0" ::: "memory");
#else
        *(uint4*)(sA[b] + soff)      = ar0;
        *(uint4*)(sA[b] + soff + 16) = ar1;
#endif
        __syncthreads();   // single barrier per K-step (triple buffering)

        v16bf af[4], bfr[2];
#pragma unroll
        for (int fm = 0; fm < 4; ++fm)
            af[fm] = ldfragA(sA[b] + (wm * 64 + fm * 16 + l16) * 64, half);
#pragma unroll
        for (int fn = 0; fn < 2; ++fn)
            bfr[fn] = ldfragB(sB[b] + (wn * 32 + fn * 16 + l16) * 64, half);
#pragma unroll
        for (int fm = 0; fm < 4; ++fm)
#pragma unroll
            for (int fn = 0; fn < 2; ++fn)
                acc[fm][fn] = wmma_bf16(af[fm], bfr[fn], acc[fm][fn]);

        aptr += 32;
        wptr += 32;
#pragma unroll
        for (int e = 0; e < 4; ++e) wr[e] = wr2[e];
#if !USE_ASYNC_LDS
        ar0 = an0; ar1 = an1;
#endif
    }

#pragma unroll
    for (int fm = 0; fm < 4; ++fm)
#pragma unroll
        for (int fn = 0; fn < 2; ++fn)
#pragma unroll
            for (int i = 0; i < 8; ++i) {
                int m = m0 + wm * 64 + fm * 16 + i + 8 * half;
                int n = n0 + wn * 32 + fn * 16 + l16;
                epi(m, n, N, acc[fm][fn][i]);
            }
}

// ---------------- RoPE + KV-cache scatter + bf16 pack ----------------
__global__ __launch_bounds__(256)
void rope_scatter_kernel(const float* __restrict__ qkv, const float* __restrict__ cosT,
                         const float* __restrict__ sinT, const int* __restrict__ slots,
                         __bf16* __restrict__ qb, __bf16* __restrict__ kb,
                         __bf16* __restrict__ vb, float* __restrict__ kcache,
                         float* __restrict__ vcache) {
    int t = blockIdx.x, tid = threadIdx.x;
    const float* q = qkv + (size_t)t * 3 * H_;
    const float* k = q + H_;
    const float* v = k + H_;
    int slot = slots[t];
#pragma unroll
    for (int j = 0; j < H_ / 256; ++j) {
        int d = tid + j * 256;
        int r = d & (HS_ - 1);
        float qv, kv;
        if (r < ROT_) {
            float cs = cosT[t * ROT_ + r], sn = sinT[t * ROT_ + r];
            qv = q[d] * cs - q[d + ROT_] * sn;
            kv = k[d] * cs - k[d + ROT_] * sn;
        } else if (r < 2 * ROT_) {
            int c = r - ROT_;
            float cs = cosT[t * ROT_ + c], sn = sinT[t * ROT_ + c];
            qv = q[d - ROT_] * sn + q[d] * cs;
            kv = k[d - ROT_] * sn + k[d] * cs;
        } else {
            qv = q[d];
            kv = k[d];
        }
        float vv = v[d];
        qb[(size_t)t * H_ + d] = (__bf16)qv;
        kb[(size_t)t * H_ + d] = (__bf16)kv;
        vb[(size_t)t * H_ + d] = (__bf16)vv;
        kcache[(size_t)slot * H_ + d] = kv;   // (NB,BS,NH,HS) flat == slot*H + d
        vcache[(size_t)slot * H_ + d] = vv;
    }
}

// ---------------- flash attention (causal), WMMA both matmuls ----------------
__global__ __launch_bounds__(128)
void attn_kernel(const __bf16* __restrict__ Q, const __bf16* __restrict__ K,
                 const __bf16* __restrict__ V, __bf16* __restrict__ O) {
    __shared__ __align__(16) char sQ[64 * 256];    // 64 q-rows x 128 bf16
    __shared__ __align__(16) char sK[32 * 256];    // 32 k-rows x 128 bf16
    __shared__ __align__(16) char sVT[128 * 64];   // V^T: 128 hs-rows x 32 keys
    __shared__ __align__(16) char sP[4 * 16 * 64]; // per-wave 16x32 P tile

    int head = blockIdx.x, q0 = blockIdx.y * 64;
    int tid = threadIdx.x, wave = tid >> 5, lane = tid & 31;
    int half = lane >> 4, l16 = lane & 15;

    // load Q tile
    for (int i = tid; i < 64 * 16; i += 128) {
        int row = i >> 4, c = i & 15;
        ((uint4*)sQ)[i] = *(const uint4*)(Q + (size_t)(q0 + row) * H_ + head * HS_ + c * 8);
    }

    v8f acc[8] = {};
    float mrow[8], lrow[8];
#pragma unroll
    for (int i = 0; i < 8; ++i) { mrow[i] = -3.0e38f; lrow[i] = 0.f; }

    const float scale = 0.08838834764831845f;  // 1/sqrt(128)
    int ktiles = q0 / 32 + 2;                  // causal bound
    for (int kt = 0; kt < ktiles; ++kt) {
        int kv0 = kt * 32;
        __syncthreads();
        for (int i = tid; i < 32 * 16; i += 128) {
            int row = i >> 4, c = i & 15;
            ((uint4*)sK)[i] = *(const uint4*)(K + (size_t)(kv0 + row) * H_ + head * HS_ + c * 8);
            union { uint4 u; __bf16 h[8]; } vv;
            vv.u = *(const uint4*)(V + (size_t)(kv0 + row) * H_ + head * HS_ + c * 8);
#pragma unroll
            for (int e = 0; e < 8; ++e)
                *((__bf16*)sVT + (size_t)(c * 8 + e) * 32 + row) = vv.h[e];
        }
        __syncthreads();

        // S = Q K^T  (16 q-rows per wave x 32 keys), contract over HS=128
        v8f s0 = {}, s1 = {};
#pragma unroll
        for (int ks = 0; ks < 4; ++ks) {
            v16bf aq  = ldfragA(sQ + (wave * 16 + l16) * 256 + ks * 64, half);
            v16bf bk0 = ldfragB(sK + (l16) * 256 + ks * 64, half);
            v16bf bk1 = ldfragB(sK + (16 + l16) * 256 + ks * 64, half);
            s0 = wmma_bf16(aq, bk0, s0);
            s1 = wmma_bf16(aq, bk1, s1);
        }

        __bf16* sp = (__bf16*)sP + wave * 16 * 32;
#pragma unroll
        for (int i = 0; i < 8; ++i) {
            int qrow = q0 + wave * 16 + i + 8 * half;
            float v0 = s0[i] * scale, v1 = s1[i] * scale;
            if (kv0 + l16 > qrow)      v0 = -3.0e38f;
            if (kv0 + 16 + l16 > qrow) v1 = -3.0e38f;
            float mx = fmaxf(v0, v1);
            for (int off = 8; off > 0; off >>= 1)
                mx = fmaxf(mx, __shfl_xor(mx, off, 32));
            float mn = fmaxf(mrow[i], mx);
            float p0 = __expf(v0 - mn), p1 = __expf(v1 - mn);
            float sum = p0 + p1;
            for (int off = 8; off > 0; off >>= 1)
                sum += __shfl_xor(sum, off, 32);
            float alpha = __expf(mrow[i] - mn);
            lrow[i] = lrow[i] * alpha + sum;
            mrow[i] = mn;
#pragma unroll
            for (int f = 0; f < 8; ++f) acc[f][i] *= alpha;
            sp[(i + 8 * half) * 32 + l16]      = (__bf16)p0;
            sp[(i + 8 * half) * 32 + 16 + l16] = (__bf16)p1;
        }
        asm volatile("s_wait_dscnt 0" ::: "memory");  // own-wave LDS RAW

        // O += P V : A = P (16x32), B = V^T columns (keys as K)
        v16bf ap = ldfragA((char*)sP + wave * 1024 + l16 * 64, half);
#pragma unroll
        for (int f = 0; f < 8; ++f) {
            v16bf bv = ldfragB(sVT + (f * 16 + l16) * 64, half);
            acc[f] = wmma_bf16(ap, bv, acc[f]);
        }
    }

#pragma unroll
    for (int f = 0; f < 8; ++f)
#pragma unroll
        for (int i = 0; i < 8; ++i) {
            int qrow = q0 + wave * 16 + i + 8 * half;
            int col = f * 16 + l16;
            O[(size_t)qrow * H_ + head * HS_ + col] = (__bf16)(acc[f][i] / lrow[i]);
        }
}

// ---------------- SiLU(gate) * up ----------------
__global__ __launch_bounds__(256)
void silu_mul_kernel(const __bf16* __restrict__ g, const __bf16* __restrict__ u,
                     __bf16* __restrict__ o, long n) {
    long i = (long)blockIdx.x * blockDim.x + threadIdx.x;
    long stride = (long)gridDim.x * blockDim.x;
    for (; i < n; i += stride) {
        float gf = (float)g[i], uf = (float)u[i];
        o[i] = (__bf16)(gf / (1.f + __expf(-gf)) * uf);
    }
}

// ---------------- launch ----------------
extern "C" void kernel_launch(void* const* d_in, const int* in_sizes, int n_in,
                              void* d_out, int out_size, void* d_ws, size_t ws_size,
                              hipStream_t stream) {
    const float* hidden    = (const float*)d_in[0];
    const float* residual  = (const float*)d_in[1];
    const float* cosT      = (const float*)d_in[2];
    const float* sinT      = (const float*)d_in[3];
    const int*   slots     = (const int*)d_in[4];
    const float* ln1_w     = (const float*)d_in[7];
    const float* w_qkv     = (const float*)d_in[8];
    const float* b_qkv     = (const float*)d_in[9];
    const float* w_dense   = (const float*)d_in[10];
    const float* ln2_w     = (const float*)d_in[11];
    const float* w_gate_up = (const float*)d_in[12];
    const float* w_down    = (const float*)d_in[13];

    float* out_mlp = (float*)d_out;
    float* out_res = out_mlp + (size_t)T_ * H_;
    float* out_kc  = out_res + (size_t)T_ * H_;
    float* out_vc  = out_kc  + (size_t)T_ * H_;

    constexpr size_t SZ_TH_F = (size_t)T_ * H_ * 4;
    constexpr size_t SZ_TH_B = (size_t)T_ * H_ * 2;
    constexpr size_t SZ_TI_B = (size_t)T_ * I_ * 2;
    char* ws = (char*)d_ws;
    float*  res     = (float*)ws;                                  size_t off = SZ_TH_F;
    __bf16* normed  = (__bf16*)(ws + off);                         off += SZ_TH_B;
    float*  qkv     = (float*)(ws + off);                          off += (size_t)T_ * 3 * H_ * 4;
    __bf16* qb      = (__bf16*)(ws + off);                         off += SZ_TH_B;
    __bf16* kb      = (__bf16*)(ws + off);                         off += SZ_TH_B;
    __bf16* vb      = (__bf16*)(ws + off);                         off += SZ_TH_B;
    __bf16* ctxb    = (__bf16*)(ws + off);                         off += SZ_TH_B;
    float*  attn_o  = (float*)(ws + off);                          off += SZ_TH_F;
    __bf16* normed2 = (__bf16*)(ws + off);                         off += SZ_TH_B;
    __bf16* gateb   = (__bf16*)(ws + off);                         off += SZ_TI_B;
    __bf16* upb     = (__bf16*)(ws + off);                         off += SZ_TI_B;
    __bf16* actb    = (__bf16*)(ws + off);

    // 1. rmsnorm(hidden + residual)
    rmsnorm_kernel<<<T_, 256, 0, stream>>>(hidden, residual, ln1_w, res, normed);
    // 2. qkv = normed @ w_qkv^T + b
    gemm_kernel<<<dim3(3 * H_ / 128, T_ / 128), 256, 0, stream>>>(
        normed, w_qkv, T_, 3 * H_, H_, EpiF32{qkv, b_qkv});
    // 3. rope + cache scatter + bf16 pack
    rope_scatter_kernel<<<T_, 256, 0, stream>>>(qkv, cosT, sinT, slots,
                                                qb, kb, vb, out_kc, out_vc);
    // 4. causal flash attention
    attn_kernel<<<dim3(NH_, T_ / 64), 128, 0, stream>>>(qb, kb, vb, ctxb);
    // 5. attn_out = ctx @ w_dense^T
    gemm_kernel<<<dim3(H_ / 128, T_ / 128), 256, 0, stream>>>(
        ctxb, w_dense, T_, H_, H_, EpiF32{attn_o, nullptr});
    // 6. rmsnorm2(attn_out + res) -> attn_res (output), normed2
    rmsnorm_kernel<<<T_, 256, 0, stream>>>(attn_o, res, ln2_w, out_res, normed2);
    // 7/8. gate and up projections
    gemm_kernel<<<dim3(I_ / 128, T_ / 128), 256, 0, stream>>>(
        normed2, w_gate_up, T_, I_, H_, EpiB16{gateb});
    gemm_kernel<<<dim3(I_ / 128, T_ / 128), 256, 0, stream>>>(
        normed2, w_gate_up + (size_t)I_ * H_, T_, I_, H_, EpiB16{upb});
    // 9. silu(gate)*up
    silu_mul_kernel<<<4096, 256, 0, stream>>>(gateb, upb, actb, (long)T_ * I_);
    // 10. mlp_out = act @ w_down^T
    gemm_kernel<<<dim3(H_ / 128, T_ / 128), 256, 0, stream>>>(
        actb, w_down, T_, H_, I_, EpiF32{out_mlp, nullptr});
}
